// LSTM_20143396618314
// MI455X (gfx1250) — matrix operand order repaired and checked by reading, not compiled, not run on hardware
//
#include <hip/hip_runtime.h>
#include <hip/hip_bf16.h>

// Problem sizes (fixed by the reference).
#define T_STEPS 50
#define B_SZ    1024
#define NP_SZ   512
#define NH_SZ   512
#define NG_SZ   4096

// Block-tiled GEMM geometry.
#define BLK      128   // block tile is BLK x BLK
#define KB       32    // K-slice per stage (one WMMA K)
#define LDS_STR  40    // 32 + 8 bf16 pad -> 20-DWORD row stride (conflict-free)

typedef __bf16 bf16;
typedef __attribute__((ext_vector_type(16))) __bf16 v16bf;
typedef __attribute__((ext_vector_type(8)))  float  v8f;
typedef __attribute__((ext_vector_type(4)))  float  v4f;

// Load a 16-element bf16 fragment as two 16-byte chunks (works for global or LDS).
__device__ __forceinline__ v16bf ld_frag(const bf16* p0, const bf16* p1) {
    union { v16bf v; v4f q[2]; } u;
    u.q[0] = *(const v4f*)p0;
    u.q[1] = *(const v4f*)p1;
    return u.v;
}

__device__ __forceinline__ float sigmoidf_fast(float x) {
    return 1.0f / (1.0f + __expf(-x));
}

// ---------------------------------------------------------------------------
// fp32 -> bf16 conversion (weights / p0).
// ---------------------------------------------------------------------------
__global__ void cvt_f32_bf16_kernel(const float* __restrict__ in,
                                    bf16* __restrict__ out, int n) {
    int i = blockIdx.x * blockDim.x + threadIdx.x;
    if (i < n) out[i] = (bf16)in[i];
}

// ---------------------------------------------------------------------------
// LDS-staged D = A(bf16, MxK) @ W(bf16, NxK)^T [+ bias] [relu], out f32/bf16.
//
// Block: 256 threads = 8 waves, block tile 128x128, double-buffered LDS
// staging of 128x32 A/B k-slices (software-pipelined global->reg->LDS).
// Wave tile: 32(M) x 64(N) = 8 accumulators; 2 A frags reused over 4 B frags.
//
// WMMA operand layouts as loaded by this kernel (bf16 16x16x32; contiguous k order):
//   A: lane m=lane&15; lanes<16 hold K0..15, lanes>=16 K16..31
//   B: lane n=lane&15; lanes<16 hold K0..15, lanes>=16 K16..31 (contig in W row)
//   C/D: vgpr r, lanes<16 -> (M=r,N=lane), lanes>=16 -> (M=r+8,N=lane-16)
// ---------------------------------------------------------------------------
__global__ __launch_bounds__(256)
void gemm_lds_kernel(const bf16* __restrict__ A, const bf16* __restrict__ W,
                     const float* __restrict__ bias,
                     float* __restrict__ outf, bf16* __restrict__ outb,
                     int M, int N, int K, int do_relu) {
    __shared__ bf16 sA[2][BLK * LDS_STR];
    __shared__ bf16 sB[2][BLK * LDS_STR];

    const int tid  = threadIdx.x;
    const int lane = tid & 31;
    const int wid  = tid >> 5;

    const int tilesN = N / BLK;
    const int bm = blockIdx.x / tilesN;
    const int bn = blockIdx.x - bm * tilesN;
    const int row0 = bm * BLK;
    const int col0 = bn * BLK;

    // Staging map: 128 rows x 64B k-slice = 512 16B chunks; 2 per thread.
    const int ar0 = tid >> 1;                 // rows 0..127 (chunk tid)
    const int aq0 = (tid & 1) * 8;            // hmm -- see below (recomputed)
    // chunk c in [0,512): row = c>>2, quarter = (c&3)*8 elements
    const int c0 = tid, c1 = tid + 256;
    const int r0 = c0 >> 2, q0 = (c0 & 3) * 8;
    const int r1 = c1 >> 2, q1 = (c1 & 3) * 8;
    (void)ar0; (void)aq0;

    const bf16* gA0 = A + (size_t)(row0 + r0) * K + q0;
    const bf16* gA1 = A + (size_t)(row0 + r1) * K + q1;
    const bf16* gB0 = W + (size_t)(col0 + r0) * K + q0;
    const bf16* gB1 = W + (size_t)(col0 + r1) * K + q1;

    // Wave tile position: 4(M) x 2(N) wave grid.
    const int wm = (wid & 3) * 32;
    const int wn = (wid >> 2) * 64;
    const int mlane = lane & 15;
    const int aoff = (lane < 16) ? 0 : 16;
    const int boff = (lane < 16) ? 0 : 16;

    v8f zero = {0.f, 0.f, 0.f, 0.f, 0.f, 0.f, 0.f, 0.f};
    v8f acc[2][4];
    #pragma unroll
    for (int i = 0; i < 2; ++i)
        #pragma unroll
        for (int j = 0; j < 4; ++j) acc[i][j] = zero;

    // Pipeline: preload k-block 0 into registers.
    v4f ra0 = *(const v4f*)gA0;
    v4f ra1 = *(const v4f*)gA1;
    v4f rb0 = *(const v4f*)gB0;
    v4f rb1 = *(const v4f*)gB1;

    const int NKB = K / KB;
    for (int kb = 0; kb < NKB; ++kb) {
        const int buf = kb & 1;
        // Stage current tile into LDS.
        *(v4f*)&sA[buf][r0 * LDS_STR + q0] = ra0;
        *(v4f*)&sA[buf][r1 * LDS_STR + q1] = ra1;
        *(v4f*)&sB[buf][r0 * LDS_STR + q0] = rb0;
        *(v4f*)&sB[buf][r1 * LDS_STR + q1] = rb1;
        // Issue next tile's global loads (hidden behind this tile's compute).
        if (kb + 1 < NKB) {
            const int k1 = (kb + 1) * KB;
            ra0 = *(const v4f*)(gA0 + k1);
            ra1 = *(const v4f*)(gA1 + k1);
            rb0 = *(const v4f*)(gB0 + k1);
            rb1 = *(const v4f*)(gB1 + k1);
        }
        // Prefetch two tiles ahead into cache (global_prefetch_b8).
        if (kb + 2 < NKB) {
            const int k2 = (kb + 2) * KB;
            __builtin_prefetch(gA0 + k2, 0, 1);
            __builtin_prefetch(gA1 + k2, 0, 1);
            __builtin_prefetch(gB0 + k2, 0, 1);
            __builtin_prefetch(gB1 + k2, 0, 1);
        }
        __syncthreads();

        const bf16* pa0 = &sA[buf][(wm + mlane) * LDS_STR + aoff];
        const bf16* pa1 = pa0 + 16 * LDS_STR;
        v16bf a0 = ld_frag(pa0, pa0 + 8);
        v16bf a1 = ld_frag(pa1, pa1 + 8);
        #pragma unroll
        for (int j = 0; j < 4; ++j) {
            const bf16* pb = &sB[buf][(wn + j * 16 + mlane) * LDS_STR + boff];
            v16bf b = ld_frag(pb, pb + 8);
            acc[0][j] = __builtin_amdgcn_wmma_f32_16x16x32_bf16(
                false, a0, false, b, (short)0, acc[0][j], false, false);
            acc[1][j] = __builtin_amdgcn_wmma_f32_16x16x32_bf16(
                false, a1, false, b, (short)0, acc[1][j], false, false);
        }
        __syncthreads();  // readers done before this buffer is overwritten (kb+2)
    }

    // Epilogue.
    const int hi = (lane >> 4) << 3;
    #pragma unroll
    for (int i = 0; i < 2; ++i) {
        #pragma unroll
        for (int j = 0; j < 4; ++j) {
            const int col = col0 + wn + j * 16 + mlane;
            const float bv = bias ? bias[col] : 0.0f;
            #pragma unroll
            for (int r = 0; r < 8; ++r) {
                const int row = row0 + wm + i * 16 + hi + r;
                float x = acc[i][j][r] + bv;
                if (do_relu) x = fmaxf(x, 0.0f);
                if (outf) outf[(size_t)row * N + col] = x;
                else      outb[(size_t)row * N + col] = (bf16)x;
            }
        }
    }
}

// ---------------------------------------------------------------------------
// One LSTM step: gates = hx@w_hh^T (+ v_t@w_ih^T + b_ih + b_hh), activations,
// cx/hx update. One wave owns a 16(batch) x 16(hidden) tile and accumulates
// all four gate blocks (i,f,g,o) so the epilogue is purely lane-local.
// hx is double-buffered by the host (read hx_in, write hx_out).
// ---------------------------------------------------------------------------
__global__ __launch_bounds__(256)
void lstm_gates_kernel(const bf16* __restrict__ hx_in,
                       bf16* __restrict__ hx_out,
                       float* __restrict__ cx,
                       const float* __restrict__ v_t,   // [B,2]
                       const float* __restrict__ w_ih,  // [4*NH,2]
                       const float* __restrict__ b_ih,  // [4*NH]
                       const float* __restrict__ b_hh,  // [4*NH]
                       const bf16* __restrict__ w_hh)   // [4*NH,NH]
{
    const int lane = threadIdx.x & 31;
    const int wave = blockIdx.x * (blockDim.x >> 5) + (threadIdx.x >> 5);
    const int tilesN = NH_SZ >> 4;  // 32
    if (wave >= (B_SZ >> 4) * tilesN) return;
    const int mt = wave / tilesN;
    const int nt = wave - mt * tilesN;
    const int row0 = mt << 4;
    const int col0 = nt << 4;

    v8f zero = {0.f, 0.f, 0.f, 0.f, 0.f, 0.f, 0.f, 0.f};
    v8f acc[4] = {zero, zero, zero, zero};

    const int mlane = lane & 15;
    const int aoff  = (lane < 16) ? 0 : 16;
    const int boff  = (lane < 16) ? 0 : 16;
    const bf16* arow = hx_in + (size_t)(row0 + mlane) * NH_SZ;
    const bf16* wr0 = w_hh + (size_t)(0 * NH_SZ + col0 + mlane) * NH_SZ + boff;
    const bf16* wr1 = w_hh + (size_t)(1 * NH_SZ + col0 + mlane) * NH_SZ + boff;
    const bf16* wr2 = w_hh + (size_t)(2 * NH_SZ + col0 + mlane) * NH_SZ + boff;
    const bf16* wr3 = w_hh + (size_t)(3 * NH_SZ + col0 + mlane) * NH_SZ + boff;

    #pragma unroll 4
    for (int k0 = 0; k0 < NH_SZ; k0 += 32) {
        v16bf a  = ld_frag(arow + k0 + aoff, arow + k0 + aoff + 8);
        v16bf b0 = ld_frag(wr0 + k0, wr0 + k0 + 8);
        v16bf b1 = ld_frag(wr1 + k0, wr1 + k0 + 8);
        v16bf b2 = ld_frag(wr2 + k0, wr2 + k0 + 8);
        v16bf b3 = ld_frag(wr3 + k0, wr3 + k0 + 8);
        acc[0] = __builtin_amdgcn_wmma_f32_16x16x32_bf16(false, a, false, b0, (short)0, acc[0], false, false);
        acc[1] = __builtin_amdgcn_wmma_f32_16x16x32_bf16(false, a, false, b1, (short)0, acc[1], false, false);
        acc[2] = __builtin_amdgcn_wmma_f32_16x16x32_bf16(false, a, false, b2, (short)0, acc[2], false, false);
        acc[3] = __builtin_amdgcn_wmma_f32_16x16x32_bf16(false, a, false, b3, (short)0, acc[3], false, false);
    }

    // Epilogue: each lane holds column j = col0 + (lane&15), rows mbase..mbase+7.
    const int j = col0 + mlane;
    float wih0[4], wih1[4], bsum[4];
    #pragma unroll
    for (int g = 0; g < 4; ++g) {
        const int jj = g * NH_SZ + j;
        wih0[g] = w_ih[jj * 2 + 0];
        wih1[g] = w_ih[jj * 2 + 1];
        bsum[g] = b_ih[jj] + b_hh[jj];
    }
    const int mbase = row0 + ((lane >> 4) << 3);
    #pragma unroll
    for (int r = 0; r < 8; ++r) {
        const int b = mbase + r;
        const float v0 = v_t[b * 2 + 0];
        const float v1 = v_t[b * 2 + 1];
        const float gi = acc[0][r] + v0 * wih0[0] + v1 * wih1[0] + bsum[0];
        const float gf = acc[1][r] + v0 * wih0[1] + v1 * wih1[1] + bsum[1];
        const float gg = acc[2][r] + v0 * wih0[2] + v1 * wih1[2] + bsum[2];
        const float go = acc[3][r] + v0 * wih0[3] + v1 * wih1[3] + bsum[3];
        const float i_ = sigmoidf_fast(gi);
        const float f_ = sigmoidf_fast(gf);
        const float g_ = tanhf(gg);
        const float o_ = sigmoidf_fast(go);
        const size_t idx = (size_t)b * NH_SZ + j;
        const float c_ = f_ * cx[idx] + i_ * g_;
        cx[idx] = c_;
        hx_out[idx] = (bf16)(o_ * tanhf(c_));
    }
}

// ---------------------------------------------------------------------------
// Host-side orchestration.
// ---------------------------------------------------------------------------
extern "C" void kernel_launch(void* const* d_in, const int* in_sizes, int n_in,
                              void* d_out, int out_size, void* d_ws, size_t ws_size,
                              hipStream_t stream) {
    const float* v      = (const float*)d_in[0];   // [T,B,2]
    const float* p0     = (const float*)d_in[1];   // [B,NP]
    const float* enc1_w = (const float*)d_in[2];   // [NH,NP]
    const float* enc1_b = (const float*)d_in[3];
    const float* enc2_w = (const float*)d_in[4];
    const float* enc2_b = (const float*)d_in[5];
    const float* w_ih   = (const float*)d_in[6];   // [4NH,2]
    const float* w_hh   = (const float*)d_in[7];   // [4NH,NH]
    const float* b_ih   = (const float*)d_in[8];
    const float* b_hh   = (const float*)d_in[9];
    const float* g_w    = (const float*)d_in[10];  // [NG,NH]
    const float* dec_w  = (const float*)d_in[11];  // [NP,NG]
    const float* dec_b  = (const float*)d_in[12];
    float* out = (float*)d_out;                    // [T,B,NP]

    // Workspace carve-out (bytes, 256-aligned).
    char* base = (char*)d_ws;
    size_t cur = 0;
    auto carve = [&](size_t bytes) -> char* {
        char* p = base + cur;
        cur = (cur + bytes + 255) & ~(size_t)255;
        return p;
    };
    bf16*  p0_bf  = (bf16*)carve((size_t)B_SZ * NP_SZ * 2);
    bf16*  e1w_bf = (bf16*)carve((size_t)NH_SZ * NP_SZ * 2);
    bf16*  e2w_bf = (bf16*)carve((size_t)NH_SZ * NP_SZ * 2);
    bf16*  whh_bf = (bf16*)carve((size_t)4 * NH_SZ * NH_SZ * 2);
    bf16*  gw_bf  = (bf16*)carve((size_t)NG_SZ * NH_SZ * 2);
    bf16*  dw_bf  = (bf16*)carve((size_t)NP_SZ * NG_SZ * 2);
    bf16*  hx0    = (bf16*)carve((size_t)B_SZ * NH_SZ * 2);
    bf16*  hx1    = (bf16*)carve((size_t)B_SZ * NH_SZ * 2);
    float* cx     = (float*)carve((size_t)B_SZ * NH_SZ * 4);
    // 4 timestep slabs of g (bf16) so the decoder runs batched (M up to 4096).
    bf16*  g_buf  = (bf16*)carve((size_t)4 * B_SZ * NG_SZ * 2);

    auto cvt = [&](const float* in, bf16* o, int n) {
        cvt_f32_bf16_kernel<<<(n + 255) / 256, 256, 0, stream>>>(in, o, n);
    };
    cvt(p0,     p0_bf,  B_SZ * NP_SZ);
    cvt(enc1_w, e1w_bf, NH_SZ * NP_SZ);
    cvt(enc2_w, e2w_bf, NH_SZ * NP_SZ);
    cvt(w_hh,   whh_bf, 4 * NH_SZ * NH_SZ);
    cvt(g_w,    gw_bf,  NG_SZ * NH_SZ);
    cvt(dec_w,  dw_bf,  NP_SZ * NG_SZ);

    auto gemm = [&](const bf16* A, const bf16* W, const float* bias,
                    float* outf, bf16* outb, int M, int N, int K, int relu) {
        int blocks = (M / BLK) * (N / BLK);
        gemm_lds_kernel<<<blocks, 256, 0, stream>>>(A, W, bias, outf, outb,
                                                    M, N, K, relu);
    };

    // Encoders: hx0 = p0@enc1_w^T + b (bf16), cx = p0@enc2_w^T + b (f32).
    gemm(p0_bf, e1w_bf, enc1_b, nullptr, hx0, B_SZ, NH_SZ, NP_SZ, 0);
    gemm(p0_bf, e2w_bf, enc2_b, cx, nullptr, B_SZ, NH_SZ, NP_SZ, 0);

    for (int t = 0; t < T_STEPS; ++t) {
        const bf16* hin = (t & 1) ? hx1 : hx0;
        bf16* hout      = (t & 1) ? hx0 : hx1;
        // Gates + state update: 2048 wave-tiles -> 256 blocks of 8 waves.
        lstm_gates_kernel<<<256, 256, 0, stream>>>(
            hin, hout, cx, v + (size_t)t * B_SZ * 2, w_ih, b_ih, b_hh, whh_bf);
        // g_t = relu(hx @ g_w^T) -> bf16 slab (L2-resident).
        gemm(hout, gw_bf, nullptr, nullptr,
             g_buf + (size_t)(t & 3) * B_SZ * NG_SZ, B_SZ, NG_SZ, NH_SZ, 1);
        // Batched decoder every 4 steps (or at the end): preds = g @ dec_w^T + b.
        if ((t & 3) == 3 || t == T_STEPS - 1) {
            const int t0 = t & ~3;
            const int rows = (t - t0 + 1) * B_SZ;   // 4096 or 2048; multiple of 128
            gemm(g_buf, dw_bf, dec_b, out + (size_t)t0 * B_SZ * NP_SZ, nullptr,
                 rows, NP_SZ, NG_SZ, 0);
        }
    }
}
